// CMLITargetLoss_77738908058355
// MI455X (gfx1250) — matrix operand order repaired and checked by reading, not compiled, not run on hardware
//
#include <hip/hip_runtime.h>

#define B_ 128
#define T_ 64
#define N_ 197
#define D_ 768
#define LDA 776  // padded LDS row stride in halfs: 388 dwords ≡ 4 (mod 64 banks)

typedef __attribute__((ext_vector_type(16))) _Float16 v16h;
typedef __attribute__((ext_vector_type(8)))  _Float16 v8h;
typedef __attribute__((ext_vector_type(4)))  _Float16 v4h;
typedef __attribute__((ext_vector_type(8)))  float    v8f;

__device__ __forceinline__ float wave_sum(float v) {
#pragma unroll
  for (int o = 16; o >= 1; o >>= 1) v += __shfl_xor(v, o, 32);
  return v;
}

__global__ void init_kernel(float* acc) {
  if (threadIdx.x < 4) acc[threadIdx.x] = 0.0f;
}

// One wave per (b,n) row: target row sumsq -> rnorm, image-MSE partials,
// and (for n==0) the CLS term ||text[b,0]-image[b,0]||^2.
__global__ __launch_bounds__(256) void rowstat_kernel(
    const float* __restrict__ image, const float* __restrict__ text,
    const float* __restrict__ target, float* __restrict__ rnorm,
    float* __restrict__ acc) {
  const int wv = threadIdx.x >> 5, lane = threadIdx.x & 31;
  const int row = blockIdx.x * 8 + wv;            // row = b*N_ + n
  const int b = row / N_, n = row % N_;
  const float4* pt = (const float4*)(target + (size_t)row * D_);
  const float4* pi = (const float4*)(image  + (size_t)row * D_);
  float ts = 0.f, ms = 0.f, cs = 0.f;
  for (int j = lane; j < D_ / 4; j += 32) {
    float4 t = pt[j], im = pi[j];
    ts += t.x * t.x + t.y * t.y + t.z * t.z + t.w * t.w;
    float dx = im.x - t.x, dy = im.y - t.y, dz = im.z - t.z, dw = im.w - t.w;
    ms += dx * dx + dy * dy + dz * dz + dw * dw;
  }
  if (n == 0) {  // CLS term: text[b,0] vs image[b,0]
    const float4* px = (const float4*)(text + (size_t)b * T_ * D_);
    for (int j = lane; j < D_ / 4; j += 32) {
      float4 x = px[j], im = pi[j];
      float dx = x.x - im.x, dy = x.y - im.y, dz = x.z - im.z, dw = x.w - im.w;
      cs += dx * dx + dy * dy + dz * dz + dw * dw;
    }
  }
  ts = wave_sum(ts); ms = wave_sum(ms); cs = wave_sum(cs);
  __shared__ float sm[8], sc[8];
  if (lane == 0) { rnorm[row] = rsqrtf(ts); sm[wv] = ms; sc[wv] = cs; }
  __syncthreads();
  if (threadIdx.x == 0) {
    float m = 0.f, c = 0.f;
#pragma unroll
    for (int i = 0; i < 8; ++i) { m += sm[i]; c += sc[i]; }
    atomicAdd(acc + 1, m);
    atomicAdd(acc + 0, c);
  }
}

// One workgroup (4 waves) per batch: f16 WMMA similarity + running argmax
// with double-buffered, software-pipelined B tiles, then fused fp32 masked
// token loss using the argmax indices.
__global__ __launch_bounds__(128) void argmax_loss_kernel(
    const float* __restrict__ text, const float* __restrict__ target,
    const int* __restrict__ pmask, const float* __restrict__ rnorm,
    float* __restrict__ acc) {
  __shared__ __align__(32) _Float16 sA[64 * LDA];      // text rows t=1..63 (row 63 zero)
  __shared__ __align__(32) _Float16 sB[2][16 * LDA];   // double-buffered target tiles
  __shared__ int sIdx[64];

  const int b = blockIdx.x;
  const int tid = threadIdx.x;
  const int wv = tid >> 5, lane = tid & 31;
  const int l15 = lane & 15, half = lane >> 4;

  const float4* tg4 = (const float4*)(target + (size_t)b * N_ * D_);
  const float* rn = rnorm + (size_t)b * N_;

  // Stage text (rows 1..63) as f16 into LDS; pad row 63 with zeros.
  const float4* tx4 = (const float4*)(text + (size_t)b * T_ * D_);
  for (int i = tid; i < 63 * 192; i += 128) {
    int m = i / 192, j = i % 192;
    float4 v = tx4[(m + 1) * 192 + j];
    v4h h = {(_Float16)v.x, (_Float16)v.y, (_Float16)v.z, (_Float16)v.w};
    *reinterpret_cast<v4h*>(&sA[m * LDA + j * 4]) = h;
  }
  for (int j = tid; j < 192; j += 128) {
    v4h z = {};
    *reinterpret_cast<v4h*>(&sA[63 * LDA + j * 4]) = z;
  }

  // Prologue: stage B tile 0 into buffer 0 (scaled by rnorm, f32->f16).
  for (int i = tid; i < 16 * 192; i += 128) {
    int nl = i / 192, j = i % 192;
    int n = 1 + nl;  // nt = 0
    float s = rn[n];
    float4 v = tg4[n * 192 + j];
    v4h h = {(_Float16)(v.x * s), (_Float16)(v.y * s),
             (_Float16)(v.z * s), (_Float16)(v.w * s)};
    *reinterpret_cast<v4h*>(&sB[0][nl * LDA + j * 4]) = h;
  }

  float best[8];
  int bn[8];
#pragma unroll
  for (int r = 0; r < 8; ++r) { best[r] = -3.0e38f; bn[r] = 1; }

  const _Float16* pA = &sA[(wv * 16 + l15) * LDA];

  for (int nt = 0; nt < 13; ++nt) {  // cols n = nt*16+1 .. nt*16+16
    __syncthreads();  // buf[nt&1] staged; buf[1-(nt&1)] free to overwrite
    const int cur = nt & 1;

    // Issue global loads for tile nt+1 into registers (latency hidden by WMMA).
    float4 vreg[3];
    float  rreg[3];
#pragma unroll
    for (int s = 0; s < 3; ++s) {
      if (nt + 1 < 13) {
        int i = tid + s * 128;
        int nl = i / 192, j = i % 192;
        int n = (nt + 1) * 16 + 1 + nl;
        int nc = (n <= 196) ? n : 0;      // safe clamped address
        vreg[s] = tg4[nc * 192 + j];
        rreg[s] = (n <= 196) ? rn[nc] : 0.0f;  // zero scale zeroes pad columns
      }
    }

    // 24-step K sweep of WMMA on the current LDS tile.
    v8f c = {};
    const _Float16* pB = &sB[cur][l15 * LDA];
#pragma unroll 4
    for (int ks = 0; ks < 24; ++ks) {  // K = 768 in steps of 32
      const int k0 = ks * 32;
      union { v16h v; v8h h[2]; } a, bm;
      // A (16x32 f16): lanes<16 row=l15 K{0..7,16..23}; lanes>=16 K{8..15,24..31}
      a.h[0] = *reinterpret_cast<const v8h*>(pA + k0 + 8 * half);
      a.h[1] = *reinterpret_cast<const v8h*>(pA + k0 + 16 + 8 * half);
      // B (32x16 f16): lane = column, lanes<16 K{0..7,16..23}; lanes>=16 K{8..15,24..31}
      bm.h[0] = *reinterpret_cast<const v8h*>(pB + k0 + 8 * half);
      bm.h[1] = *reinterpret_cast<const v8h*>(pB + k0 + 16 + 8 * half);
      c = __builtin_amdgcn_wmma_f32_16x16x32_f16(false, a.v, false, bm.v,
                                                 (short)0, c, false, false);
    }

    // Running per-(row, col-slot) argmax update.
    const int n = nt * 16 + 1 + l15;  // this lane's column
    if (n <= 196) {
#pragma unroll
      for (int r = 0; r < 8; ++r) {  // VGPR r: row = r + 8*half of this M-tile
        float v = c[r];
        if (v > best[r]) { best[r] = v; bn[r] = n; }
      }
    }

    // Drain staged loads into the alternate buffer.
    if (nt + 1 < 13) {
#pragma unroll
      for (int s = 0; s < 3; ++s) {
        int i = tid + s * 128;
        int nl = i / 192, j = i % 192;
        float sc = rreg[s];
        float4 v = vreg[s];
        v4h h = {(_Float16)(v.x * sc), (_Float16)(v.y * sc),
                 (_Float16)(v.z * sc), (_Float16)(v.w * sc)};
        *reinterpret_cast<v4h*>(&sB[1 - cur][nl * LDA + j * 4]) = h;
      }
    }
  }

  // Reduce (max, first-index tie-break) across the 16 lanes of each half.
#pragma unroll
  for (int off = 8; off >= 1; off >>= 1) {
#pragma unroll
    for (int r = 0; r < 8; ++r) {
      float ov = __shfl_xor(best[r], off, 16);
      int on = __shfl_xor(bn[r], off, 16);
      if (ov > best[r] || (ov == best[r] && on < bn[r])) { best[r] = ov; bn[r] = on; }
    }
  }
  if (l15 == 0) {
#pragma unroll
    for (int r = 0; r < 8; ++r) sIdx[wv * 16 + half * 8 + r] = bn[r];
  }
  __syncthreads();

  // Fused masked token loss in fp32: m = t-1, aligned = target[b, sIdx[m]].
  float wacc = 0.f, wkeep = 0.f;
  for (int m = wv; m < 63; m += 4) {
    const int t = m + 1;
    const int idx = sIdx[m];
    const float4* pt = (const float4*)(text + ((size_t)b * T_ + t) * D_);
    const float4* pa = (const float4*)(target + ((size_t)b * N_ + idx) * D_);
    float s = 0.f;
    for (int j = lane; j < D_ / 4; j += 32) {
      float4 x = pt[j], y = pa[j];
      float dx = x.x - y.x, dy = x.y - y.y, dz = x.z - y.z, dw = x.w - y.w;
      s += dx * dx + dy * dy + dz * dz + dw * dw;
    }
    s = wave_sum(s);
    if (lane == 0 && pmask[b * T_ + t] == 0) { wacc += s; wkeep += 1.f; }
  }
  if (lane == 0) {
    atomicAdd(acc + 0, wacc);
    atomicAdd(acc + 2, wkeep);
  }
}

__global__ void finalize_kernel(const float* __restrict__ acc, float* __restrict__ out) {
  if (threadIdx.x == 0) {
    float text_loss = acc[0] / ((128.0f + acc[2]) * 768.0f);
    float img_loss = acc[1] / (128.0f * 197.0f * 768.0f);
    out[0] = 0.5f * (text_loss + img_loss);
  }
}

extern "C" void kernel_launch(void* const* d_in, const int* in_sizes, int n_in,
                              void* d_out, int out_size, void* d_ws, size_t ws_size,
                              hipStream_t stream) {
  const float* image = (const float*)d_in[0];
  const float* text = (const float*)d_in[1];
  const float* target = (const float*)d_in[2];
  const int* pmask = (const int*)d_in[3];
  float* out = (float*)d_out;
  float* acc = (float*)d_ws;          // acc[0]=cls+masked, acc[1]=imgMSE, acc[2]=keep
  float* rnorm = acc + 16;            // B_*N_ floats

  init_kernel<<<1, 32, 0, stream>>>(acc);
  rowstat_kernel<<<(B_ * N_) / 8, 256, 0, stream>>>(image, text, target, rnorm, acc);
  argmax_loss_kernel<<<B_, 128, 0, stream>>>(text, target, pmask, rnorm, acc);
  finalize_kernel<<<1, 1, 0, stream>>>(acc, out);
}
